// LSTMPredictor_76647986364497
// MI455X (gfx1250) — matrix operand order repaired and checked by reading, not compiled, not run on hardware
//
#include <hip/hip_runtime.h>

typedef __bf16 bf16_t;
typedef __attribute__((ext_vector_type(16))) __bf16 v16bf;
typedef __attribute__((ext_vector_type(8)))  __bf16 v8bf;
typedef __attribute__((ext_vector_type(8)))  float  v8f;

#define BB  1024
#define TT  256
#define DIN 6
#define DPAD 32   // DIN padded to one WMMA K-block
#define HH  128
#define G4  512   // 4*H

// ---------------- helpers ----------------

// fast sigmoid / tanh on the v_exp_f32 + v_rcp_f32 trans path
__device__ __forceinline__ float sigf(float x) {
  return __builtin_amdgcn_rcpf(1.0f + __expf(-x));
}
__device__ __forceinline__ float tanh_fast(float x) {
  float e = __expf(-2.0f * fabsf(x));
  float r = (1.0f - e) * __builtin_amdgcn_rcpf(1.0f + e);
  return copysignf(r, x);
}

// A-fragment (16x32 bf16): lanes 0-15 hold M=lane, K = {0..15};
// lanes 16-31 hold M=lane-16, K = {16..31}. `row` points at this lane's
// matrix row with the k-block base already applied.
__device__ __forceinline__ v16bf load_a_frag(const bf16_t* row, int lane) {
  const int hi = (lane >> 4) & 1;
  v8bf a0 = *reinterpret_cast<const v8bf*>(row + hi * 16);
  v8bf a1 = *reinterpret_cast<const v8bf*>(row + hi * 16 + 8);
  return __builtin_shufflevector(a0, a1, 0,1,2,3,4,5,6,7,8,9,10,11,12,13,14,15);
}

// B-fragment (32x16 bf16): lane (0-15) = column N, holds K = {0..15}; lanes
// 16-31 hold K = {16..31}. Weights stored row-major W[N][K] -> one 32B load.
__device__ __forceinline__ v16bf load_b_frag(const bf16_t* wrow_kbase, int lane) {
  return *reinterpret_cast<const v16bf*>(wrow_kbase + ((lane < 16) ? 0 : 16));
}

__device__ __forceinline__ v8f wmma_bf16(v16bf a, v16bf b, v8f c) {
  return __builtin_amdgcn_wmma_f32_16x16x32_bf16(false, a, false, b, (short)0, c, false, false);
}

// ---------------- tiny prep kernels ----------------

__global__ __launch_bounds__(256) void k_cvt_bf16(const float* __restrict__ src,
                                                  bf16_t* __restrict__ dst, int n) {
  int i = blockIdx.x * 256 + threadIdx.x;
  if (i < n) dst[i] = (bf16_t)src[i];
}

// Wih0 [2,512,6] fp32 -> zero-padded [2,512,32] bf16 so the layer-0 input
// projection runs on the WMMA path instead of VALU.
__global__ __launch_bounds__(256) void k_pad_wih0(const float* __restrict__ src,
                                                  bf16_t* __restrict__ dst, int n) {
  int i = blockIdx.x * 256 + threadIdx.x;
  if (i >= n) return;
  int row = i / DPAD, d = i % DPAD;
  dst[i] = (d < DIN) ? (bf16_t)src[row * DIN + d] : (bf16_t)0.f;
}

__global__ __launch_bounds__(256) void k_ln(const float* __restrict__ x,
                                            const float* __restrict__ g,
                                            const float* __restrict__ b,
                                            float* __restrict__ xo, int nrows) {
  int i = blockIdx.x * 256 + threadIdx.x;
  if (i >= nrows) return;
  const float* xr = x + (size_t)i * DIN;
  float mu = 0.f;
#pragma unroll
  for (int d = 0; d < DIN; ++d) mu += xr[d];
  mu *= (1.0f / DIN);
  float var = 0.f;
#pragma unroll
  for (int d = 0; d < DIN; ++d) { float t = xr[d] - mu; var += t * t; }
  var *= (1.0f / DIN);
  float rs = rsqrtf(var + 1e-5f);
  float* xrow = xo + (size_t)i * DIN;
#pragma unroll
  for (int d = 0; d < DIN; ++d) xrow[d] = (xr[d] - mu) * rs * g[d] + b[d];
}

// ---------------- layer-0 BiLSTM (x-proj on WMMA via padded K=32) ----------------
// grid: (B/16, 2 directions), block: 256 (8 waves)

__global__ __launch_bounds__(256) void k_lstm0(
    const float* __restrict__ xln,      // [B,T,6]
    const bf16_t* __restrict__ Wih_p,   // [2,512,32] bf16 zero-padded
    const bf16_t* __restrict__ Whh_bf,  // [2,512,128] bf16
    const float* __restrict__ bih,      // [2,512]
    const float* __restrict__ bhh,      // [2,512]
    bf16_t* __restrict__ hout)          // [B,T,256] fwd->0..127, bwd->128..255
{
  const int dir  = blockIdx.y;
  const int b0   = blockIdx.x * 16;
  const int tid  = threadIdx.x;
  const int lane = tid & 31;
  const int wv   = tid >> 5;

  __shared__ float  s_gates[16][G4];   // 32 KB
  __shared__ bf16_t s_h[16][HH];       // 4 KB
  __shared__ bf16_t s_xbf[16][DPAD];   // 1 KB, zero-padded x tile

  const bf16_t* Wih = Wih_p + (size_t)dir * G4 * DPAD;
  const bf16_t* Whh = Whh_bf + (size_t)dir * G4 * HH;

  for (int i = tid; i < 16 * HH; i += 256) (&s_h[0][0])[i] = (bf16_t)0.f;
  for (int i = tid; i < 16 * DPAD; i += 256) (&s_xbf[0][0])[i] = (bf16_t)0.f;

  // per-lane bias (bih+bhh), fixed for the whole sequence -> registers
  float bias_q[4];
#pragma unroll
  for (int q = 0; q < 4; ++q) {
    const int nn = (wv * 4 + q) * 16 + (lane & 15);
    bias_q[q] = bih[dir * G4 + nn] + bhh[dir * G4 + nn];
  }
  float c_reg[8];
#pragma unroll
  for (int i = 0; i < 8; ++i) c_reg[i] = 0.f;
  __syncthreads();

  const int mhalf = (lane < 16) ? 0 : 8;

  for (int step = 0; step < TT; ++step) {
    const int t = dir ? (TT - 1 - step) : step;
    // stage x tile (only the first 6 of 32 columns are live; rest stay zero)
    if (tid < 16 * DIN) {
      int r = tid / DIN, d = tid % DIN;
      s_xbf[r][d] = (bf16_t)xln[((size_t)(b0 + r) * TT + t) * DIN + d];
    }
    __syncthreads();

#pragma unroll
    for (int q = 0; q < 4; ++q) {
      const int nn = (wv * 4 + q) * 16 + (lane & 15);
      v8f acc;
#pragma unroll
      for (int r = 0; r < 8; ++r) acc[r] = bias_q[q];
      // input projection: one WMMA (K = 32 padded)
      {
        v16bf af = load_a_frag(&s_xbf[lane & 15][0], lane);
        v16bf bf = load_b_frag(Wih + (size_t)nn * DPAD, lane);
        acc = wmma_bf16(af, bf, acc);
      }
      // recurrence: K = 128
#pragma unroll
      for (int kb = 0; kb < HH / 32; ++kb) {
        v16bf af = load_a_frag(&s_h[lane & 15][kb * 32], lane);
        v16bf bf = load_b_frag(Whh + (size_t)nn * HH + kb * 32, lane);
        acc = wmma_bf16(af, bf, acc);
      }
#pragma unroll
      for (int r = 0; r < 8; ++r) s_gates[r + mhalf][nn] = acc[r];
    }
    __syncthreads();

#pragma unroll
    for (int i = 0; i < 8; ++i) {
      const int idx = tid * 8 + i;
      const int bl = idx >> 7, j = idx & 127;
      float gi = s_gates[bl][j];
      float gf = s_gates[bl][HH + j];
      float gg = s_gates[bl][2 * HH + j];
      float go = s_gates[bl][3 * HH + j];
      float c = sigf(gf) * c_reg[i] + sigf(gi) * tanh_fast(gg);
      c_reg[i] = c;
      float h = sigf(go) * tanh_fast(c);
      bf16_t hb = (bf16_t)h;
      s_h[bl][j] = hb;
      hout[((size_t)(b0 + bl) * TT + t) * 256 + dir * HH + j] = hb;
    }
    __syncthreads();
  }
}

// ---------------- layer-1 BiLSTM (K = 256 x-proj + 128 recurrent) ----------------

__global__ __launch_bounds__(256) void k_lstm1(
    const bf16_t* __restrict__ h0,      // [B,T,256] bf16
    const bf16_t* __restrict__ Wih_bf,  // [2,512,256] bf16
    const bf16_t* __restrict__ Whh_bf,  // [2,512,128] bf16
    const float* __restrict__ bih,      // [2,512]
    const float* __restrict__ bhh,      // [2,512]
    bf16_t* __restrict__ h1)            // [B,T,256]
{
  const int dir  = blockIdx.y;
  const int b0   = blockIdx.x * 16;
  const int tid  = threadIdx.x;
  const int lane = tid & 31;
  const int wv   = tid >> 5;

  __shared__ float  s_gates[16][G4];  // 32 KB
  __shared__ bf16_t s_h[16][HH];      // 4 KB

  const bf16_t* Wih = Wih_bf + (size_t)dir * G4 * 256;
  const bf16_t* Whh = Whh_bf + (size_t)dir * G4 * HH;

  for (int i = tid; i < 16 * HH; i += 256) (&s_h[0][0])[i] = (bf16_t)0.f;

  float bias_q[4];
#pragma unroll
  for (int q = 0; q < 4; ++q) {
    const int nn = (wv * 4 + q) * 16 + (lane & 15);
    bias_q[q] = bih[dir * G4 + nn] + bhh[dir * G4 + nn];
  }
  float c_reg[8];
#pragma unroll
  for (int i = 0; i < 8; ++i) c_reg[i] = 0.f;
  __syncthreads();

  const int mhalf = (lane < 16) ? 0 : 8;
  const bf16_t* xrow = h0 + ((size_t)(b0 + (lane & 15)) * TT) * 256;

  for (int step = 0; step < TT; ++step) {
    const int t = dir ? (TT - 1 - step) : step;
    const bf16_t* xrt = xrow + (size_t)t * 256;

    // prefetch next timestep's x row while this step's WMMAs run
    if (step + 1 < TT) {
      const int tn = dir ? (TT - 2 - step) : (step + 1);
      __builtin_prefetch(xrow + (size_t)tn * 256, 0, 0);
    }

#pragma unroll
    for (int q = 0; q < 4; ++q) {
      const int nn = (wv * 4 + q) * 16 + (lane & 15);
      v8f acc;
#pragma unroll
      for (int r = 0; r < 8; ++r) acc[r] = bias_q[q];
      // input projection: K = 256
#pragma unroll
      for (int kb = 0; kb < 256 / 32; ++kb) {
        v16bf af = load_a_frag(xrt + kb * 32, lane);
        v16bf bf = load_b_frag(Wih + (size_t)nn * 256 + kb * 32, lane);
        acc = wmma_bf16(af, bf, acc);
      }
      // recurrence: K = 128
#pragma unroll
      for (int kb = 0; kb < HH / 32; ++kb) {
        v16bf af = load_a_frag(&s_h[lane & 15][kb * 32], lane);
        v16bf bf = load_b_frag(Whh + (size_t)nn * HH + kb * 32, lane);
        acc = wmma_bf16(af, bf, acc);
      }
#pragma unroll
      for (int r = 0; r < 8; ++r) s_gates[r + mhalf][nn] = acc[r];
    }
    __syncthreads();

#pragma unroll
    for (int i = 0; i < 8; ++i) {
      const int idx = tid * 8 + i;
      const int bl = idx >> 7, j = idx & 127;
      float gi = s_gates[bl][j];
      float gf = s_gates[bl][HH + j];
      float gg = s_gates[bl][2 * HH + j];
      float go = s_gates[bl][3 * HH + j];
      float c = sigf(gf) * c_reg[i] + sigf(gi) * tanh_fast(gg);
      c_reg[i] = c;
      float h = sigf(go) * tanh_fast(c);
      bf16_t hb = (bf16_t)h;
      s_h[bl][j] = hb;
      h1[((size_t)(b0 + bl) * TT + t) * 256 + dir * HH + j] = hb;
    }
    __syncthreads();
  }
}

// ---------------- attention scores: s = tanh(h1 Wa1^T + ba1) Wa2^T + ba2 ----------------
// one wave handles 16 rows of [B*T, 256]; 8 waves/block -> 128 rows/block

__global__ __launch_bounds__(256) void k_attn(
    const bf16_t* __restrict__ h1,     // [B*T, 256]
    const bf16_t* __restrict__ Wa1b,   // [128, 256] bf16
    const float* __restrict__ ba1,     // [128]
    const float* __restrict__ Wa2,     // [128]
    const float* __restrict__ ba2,     // [1]
    float* __restrict__ scores)        // [B*T]
{
  const int tid  = threadIdx.x;
  const int lane = tid & 31;
  const int wv   = tid >> 5;
  __shared__ float s_red[128];

  for (int i = tid; i < 128; i += 256) s_red[i] = 0.f;
  __syncthreads();

  const int row0 = blockIdx.x * 128 + wv * 16;
  const bf16_t* arow = h1 + (size_t)(row0 + (lane & 15)) * 256;
  const int mhalf = (lane < 16) ? 0 : 8;

  float p[8];
#pragma unroll
  for (int r = 0; r < 8; ++r) p[r] = 0.f;

#pragma unroll
  for (int nt = 0; nt < 8; ++nt) {
    const int nn = nt * 16 + (lane & 15);
    v8f acc;
#pragma unroll
    for (int r = 0; r < 8; ++r) acc[r] = 0.f;
#pragma unroll
    for (int kb = 0; kb < 8; ++kb) {
      v16bf af = load_a_frag(arow + kb * 32, lane);
      v16bf bf = load_b_frag(Wa1b + (size_t)nn * 256 + kb * 32, lane);
      acc = wmma_bf16(af, bf, acc);
    }
    const float bb = ba1[nn], ww = Wa2[nn];
#pragma unroll
    for (int r = 0; r < 8; ++r) p[r] += tanh_fast(acc[r] + bb) * ww;
  }
#pragma unroll
  for (int r = 0; r < 8; ++r) atomicAdd(&s_red[wv * 16 + r + mhalf], p[r]);
  __syncthreads();

  if (tid < 128) scores[(size_t)blockIdx.x * 128 + tid] = s_red[tid] + ba2[0];
}

// ---------------- softmax over T + context + MLP head ----------------
// one block per batch element

__global__ __launch_bounds__(256) void k_final(
    const float* __restrict__ scores,  // [B,T]
    const bf16_t* __restrict__ h1,     // [B,T,256]
    const float* __restrict__ Wf1,     // [128,256]
    const float* __restrict__ bf1,     // [128]
    const float* __restrict__ ln2g, const float* __restrict__ ln2b,
    const float* __restrict__ Wf2,     // [5,128]
    const float* __restrict__ bf2,     // [5]
    float* __restrict__ out)           // [B,5]
{
  const int b = blockIdx.x, tid = threadIdx.x;
  __shared__ float s_attn[256];
  __shared__ float s_ctx[256];
  __shared__ float s_z[128];
  __shared__ float s_red[256];

  // softmax over T=256
  float sc = scores[(size_t)b * TT + tid];
  s_red[tid] = sc; __syncthreads();
  for (int o = 128; o > 0; o >>= 1) { if (tid < o) s_red[tid] = fmaxf(s_red[tid], s_red[tid + o]); __syncthreads(); }
  float mx = s_red[0]; __syncthreads();
  float e = __expf(sc - mx);
  s_red[tid] = e; __syncthreads();
  for (int o = 128; o > 0; o >>= 1) { if (tid < o) s_red[tid] += s_red[tid + o]; __syncthreads(); }
  float inv = 1.0f / s_red[0];
  s_attn[tid] = e * inv; __syncthreads();

  // context[c] = sum_t attn[t] * h1[b,t,c]
  const bf16_t* hb = h1 + (size_t)b * TT * 256 + tid;
  float ctx = 0.f;
  for (int t = 0; t < TT; ++t) ctx += s_attn[t] * (float)hb[(size_t)t * 256];
  s_ctx[tid] = ctx; __syncthreads();

  // y = Wf1 @ ctx + bf1 ; LN over 128 ; relu
  float y = 0.f;
  if (tid < 128) {
    const float* w = Wf1 + (size_t)tid * 256;
    for (int k = 0; k < 256; ++k) y += w[k] * s_ctx[k];
    y += bf1[tid];
  }
  s_red[tid] = (tid < 128) ? y : 0.f; __syncthreads();
  for (int o = 128; o > 0; o >>= 1) { if (tid < o) s_red[tid] += s_red[tid + o]; __syncthreads(); }
  float mu = s_red[0] * (1.0f / 128.0f); __syncthreads();
  float d2 = (tid < 128) ? (y - mu) * (y - mu) : 0.f;
  s_red[tid] = d2; __syncthreads();
  for (int o = 128; o > 0; o >>= 1) { if (tid < o) s_red[tid] += s_red[tid + o]; __syncthreads(); }
  float var = s_red[0] * (1.0f / 128.0f); __syncthreads();
  if (tid < 128) {
    float z = (y - mu) * rsqrtf(var + 1e-5f) * ln2g[tid] + ln2b[tid];
    s_z[tid] = fmaxf(z, 0.f);
  }
  __syncthreads();

  if (tid < 5) {
    const float* w = Wf2 + (size_t)tid * 128;
    float o2 = bf2[tid];
    for (int k = 0; k < 128; ++k) o2 += w[k] * s_z[k];
    out[(size_t)b * 5 + tid] = o2;
  }
}

// ---------------- host launcher ----------------

extern "C" void kernel_launch(void* const* d_in, const int* in_sizes, int n_in,
                              void* d_out, int out_size, void* d_ws, size_t ws_size,
                              hipStream_t stream) {
  (void)in_sizes; (void)n_in; (void)out_size; (void)ws_size;
  const float* x    = (const float*)d_in[0];
  const float* ln_g = (const float*)d_in[1];
  const float* ln_b = (const float*)d_in[2];
  const float* Wih0 = (const float*)d_in[3];
  const float* Whh0 = (const float*)d_in[4];
  const float* bih0 = (const float*)d_in[5];
  const float* bhh0 = (const float*)d_in[6];
  const float* Wih1 = (const float*)d_in[7];
  const float* Whh1 = (const float*)d_in[8];
  const float* bih1 = (const float*)d_in[9];
  const float* bhh1 = (const float*)d_in[10];
  const float* Wa1  = (const float*)d_in[11];
  const float* ba1  = (const float*)d_in[12];
  const float* Wa2  = (const float*)d_in[13];
  const float* ba2  = (const float*)d_in[14];
  const float* Wf1  = (const float*)d_in[15];
  const float* bf1  = (const float*)d_in[16];
  const float* ln2g = (const float*)d_in[17];
  const float* ln2b = (const float*)d_in[18];
  const float* Wf2  = (const float*)d_in[19];
  const float* bf2  = (const float*)d_in[20];
  float* out = (float*)d_out;

  char* ws = (char*)d_ws;
  size_t off = 0;
  auto salloc = [&](size_t bytes) { void* p = ws + off; off += (bytes + 255) & ~(size_t)255; return p; };
  float*  xln   = (float*) salloc((size_t)BB * TT * DIN * 4);
  bf16_t* h0    = (bf16_t*)salloc((size_t)BB * TT * 256 * 2);
  bf16_t* h1b   = (bf16_t*)salloc((size_t)BB * TT * 256 * 2);
  float*  sc    = (float*) salloc((size_t)BB * TT * 4);
  bf16_t* Whh0b = (bf16_t*)salloc((size_t)2 * G4 * HH * 2);
  bf16_t* Whh1b = (bf16_t*)salloc((size_t)2 * G4 * HH * 2);
  bf16_t* Wih1b = (bf16_t*)salloc((size_t)2 * G4 * 256 * 2);
  bf16_t* Wa1b  = (bf16_t*)salloc((size_t)HH * 256 * 2);
  bf16_t* Wih0p = (bf16_t*)salloc((size_t)2 * G4 * DPAD * 2);

  // weight conversions (tiny, L2-resident afterwards)
  {
    int n;
    n = 2 * G4 * HH;   k_cvt_bf16<<<(n + 255) / 256, 256, 0, stream>>>(Whh0, Whh0b, n);
    n = 2 * G4 * HH;   k_cvt_bf16<<<(n + 255) / 256, 256, 0, stream>>>(Whh1, Whh1b, n);
    n = 2 * G4 * 256;  k_cvt_bf16<<<(n + 255) / 256, 256, 0, stream>>>(Wih1, Wih1b, n);
    n = HH * 256;      k_cvt_bf16<<<(n + 255) / 256, 256, 0, stream>>>(Wa1, Wa1b, n);
    n = 2 * G4 * DPAD; k_pad_wih0<<<(n + 255) / 256, 256, 0, stream>>>(Wih0, Wih0p, n);
  }

  k_ln<<<(BB * TT + 255) / 256, 256, 0, stream>>>(x, ln_g, ln_b, xln, BB * TT);

  k_lstm0<<<dim3(BB / 16, 2), 256, 0, stream>>>(xln, Wih0p, Whh0b, bih0, bhh0, h0);
  k_lstm1<<<dim3(BB / 16, 2), 256, 0, stream>>>(h0, Wih1b, Whh1b, bih1, bhh1, h1b);

  k_attn<<<(BB * TT) / 128, 256, 0, stream>>>(h1b, Wa1b, ba1, Wa2, ba2, sc);
  k_final<<<BB, 256, 0, stream>>>(sc, h1b, Wf1, bf1, ln2g, ln2b, Wf2, bf2, out);
}
